// GraphormerLayer_6382321402206
// MI455X (gfx1250) — compile-verified, matrix operand order checked by reading, not run on hardware
//
#include <hip/hip_runtime.h>
#include <hip/hip_bf16.h>

// ---------------------------------------------------------------------------
// Graphormer layer for MI455X (gfx1250, wave32, WMMA f32_16x16x32_f16)
// B=2, N=2048, EMBED=256, HEADS=8, HEAD_DIM=32, FFN=1024
// GEMMs: 4 waves/block, each wave computes a 16x64 tile (4 accumulators)
// ---------------------------------------------------------------------------

#define SEQ 2048
#define BATCH 2
#define EMBED 256
#define HEADS 8
#define HDIM 32
#define FFN 1024
#define ROWS (BATCH * SEQ)   // 4096

typedef __attribute__((ext_vector_type(16))) _Float16 v16h;
typedef __attribute__((ext_vector_type(8)))  _Float16 v8h;
typedef __attribute__((ext_vector_type(8)))  float    v8f;

// Load one 16x32 f16 fragment (A-layout / mirrored B-layout) from row-major
// storage with leading dimension ld. Lane<16: row=lane, K={0..7,16..23};
// lane>=16: row=lane-16, K={8..15,24..31}. Two b128 loads per lane.
__device__ __forceinline__ v16h load_frag16x32(const _Float16* __restrict__ base, int ld) {
    const int lane = threadIdx.x & 31;
    const int row  = lane & 15;
    const int hi   = lane >> 4;
    const _Float16* p = base + (size_t)row * ld + hi * 8;
    v8h lo = *(const v8h*)(p);
    v8h hv = *(const v8h*)(p + 16);
    v16h r;
#pragma unroll
    for (int i = 0; i < 8; ++i) { r[i] = lo[i]; r[i + 8] = hv[i]; }
    return r;
}

// ---------------------------------------------------------------------------
// Weight prep: f32 [K][N] row-major -> f16 [N][K] (transposed)
// ---------------------------------------------------------------------------
__global__ void transpose_to_f16_kernel(const float* __restrict__ in,
                                        _Float16* __restrict__ out, int K, int N) {
    int idx = blockIdx.x * blockDim.x + threadIdx.x;
    if (idx >= K * N) return;
    int k = idx / N, n = idx % N;
    out[(size_t)n * K + k] = (_Float16)in[idx];
}

// ---------------------------------------------------------------------------
// LayerNorm over 256-wide rows: one wave per row, f16 output
// ---------------------------------------------------------------------------
__global__ void __launch_bounds__(32)
layernorm_kernel(const float* __restrict__ x, const float* __restrict__ w,
                 const float* __restrict__ b, _Float16* __restrict__ out) {
    const int row  = blockIdx.x;
    const int lane = threadIdx.x;
    const float* xr = x + (size_t)row * EMBED;
    float vals[8], s = 0.f, s2 = 0.f;
#pragma unroll
    for (int i = 0; i < 8; ++i) {
        float v = xr[lane + 32 * i];
        vals[i] = v; s += v; s2 += v * v;
    }
#pragma unroll
    for (int off = 16; off > 0; off >>= 1) {
        s  += __shfl_xor(s,  off, 32);
        s2 += __shfl_xor(s2, off, 32);
    }
    const float mu  = s * (1.0f / EMBED);
    const float var = s2 * (1.0f / EMBED) - mu * mu;
    const float rs  = rsqrtf(var + 1e-5f);
#pragma unroll
    for (int i = 0; i < 8; ++i) {
        const int c = lane + 32 * i;
        out[(size_t)row * EMBED + c] = (_Float16)((vals[i] - mu) * rs * w[c] + b[c]);
    }
}

// ---------------------------------------------------------------------------
// Register-blocked WMMA GEMM: block = 64x64 macro-tile, 4 waves, each wave
// computes 16x64 (4 independent accumulators; A fragment reused 4x).
// C = A[M,K] * Bt[N,K]^T + bias
// flags: 1 = exact-erf GELU, 2 = f32 output, 4 = store transposed f16 (for V)
// residual != nullptr => add residual[M,N] (f32)
// ---------------------------------------------------------------------------
__global__ void __launch_bounds__(128)
wmma_gemm_kernel(const _Float16* __restrict__ A, const _Float16* __restrict__ Bt,
                 const float* __restrict__ bias, const float* __restrict__ residual,
                 void* __restrict__ out, int M, int N, int K, int flags) {
    const int wave = threadIdx.x >> 5;         // 0..3 -> M sub-tile
    const int lane = threadIdx.x & 31;
    const int col  = lane & 15;
    const int hi   = lane >> 4;

    const int nblk = N >> 6;                   // 64-wide column blocks
    const int mb   = blockIdx.x / nblk;
    const int nb   = blockIdx.x % nblk;
    const int mt   = mb * 4 + wave;            // 16-row tile index

    const _Float16* arow  = A  + (size_t)(mt * 16) * K;
    const _Float16* bbase = Bt + (size_t)(nb * 64) * K;

    v8f acc[4] = {v8f{}, v8f{}, v8f{}, v8f{}};

    for (int kk = 0; kk < K; kk += 32) {
        // prefetch next k-slab (speculative; OOB prefetch is dropped)
        __builtin_prefetch(arow + kk + 64, 0, 1);
        __builtin_prefetch(bbase + kk + 64, 0, 1);

        v16h af = load_frag16x32(arow + kk, K);
#pragma unroll
        for (int j = 0; j < 4; ++j) {
            v16h bf = load_frag16x32(bbase + (size_t)(j * 16) * K + kk, K);
            acc[j] = __builtin_amdgcn_wmma_f32_16x16x32_f16(false, af, false, bf,
                                                            (short)0, acc[j],
                                                            false, false);
        }
    }

#pragma unroll
    for (int j = 0; j < 4; ++j) {
        const int colg = nb * 64 + j * 16 + col;
        const float bv = bias ? bias[colg] : 0.0f;
#pragma unroll
        for (int r = 0; r < 8; ++r) {
            const int rowg = mt * 16 + r + (hi << 3);
            float v = acc[j][r] + bv;
            if (flags & 1) v = 0.5f * v * (1.0f + erff(v * 0.70710678118654752f));
            if (residual)  v += residual[(size_t)rowg * N + colg];
            if (flags & 2) {
                ((float*)out)[(size_t)rowg * N + colg] = v;
            } else if (flags & 4) {
                // per-batch transposed store: vT[b][colg][tok]
                const int bb  = rowg >> 11;       // / SEQ
                const int tok = rowg & (SEQ - 1);
                ((_Float16*)out)[((size_t)(bb << 8) + colg) * SEQ + tok] = (_Float16)v;
            } else {
                ((_Float16*)out)[(size_t)rowg * N + colg] = (_Float16)v;
            }
        }
    }
}

// ---------------------------------------------------------------------------
// Flash-style attention with multiplicative influence bias.
// One wave per (b, h, 16-query tile). Streams keys in chunks of 32.
// q,k: [B,SEQ,EMBED] f16 ; vT: [B,EMBED,SEQ] f16 ; infl: [B,SEQ,SEQ] f32
// out: [B,SEQ,EMBED] f16
// ---------------------------------------------------------------------------
__global__ void __launch_bounds__(32)
attention_kernel(const _Float16* __restrict__ q, const _Float16* __restrict__ k,
                 const _Float16* __restrict__ vT, const float* __restrict__ infl,
                 _Float16* __restrict__ out) {
    __shared__ _Float16 lds_p[16 * 32];      // P tile, row-major 16x32

    const int nqt = SEQ / 16;                // 128
    const int b   = blockIdx.x / (HEADS * nqt);
    const int h   = (blockIdx.x / nqt) % HEADS;
    const int qt  = blockIdx.x % nqt;
    const int lane = threadIdx.x & 31;
    const int col  = lane & 15;
    const int hi   = lane >> 4;

    const v16h qf = load_frag16x32(
        q + ((size_t)(b * SEQ + qt * 16)) * EMBED + h * HDIM, EMBED);

    v8f acc0 = {}, acc1 = {};
    float m[8], l[8];
#pragma unroll
    for (int r = 0; r < 8; ++r) { m[r] = -3.0e38f; l[r] = 0.0f; }

    const float scale = 0.17677669529663687f;   // 1/sqrt(32)
    const float* infl_b = infl + (size_t)b * SEQ * SEQ;
    const _Float16* kbase = k  + (size_t)b * SEQ * EMBED + h * HDIM;
    const _Float16* vbase = vT + ((size_t)b * EMBED + h * HDIM) * SEQ;

    for (int kb = 0; kb < SEQ; kb += 32) {
        // prefetch next chunk of K and V while computing this one
        __builtin_prefetch(kbase + (size_t)(kb + 32) * EMBED, 0, 1);
        __builtin_prefetch(vbase + kb + 32, 0, 1);

        // ---- scores: two 16x16 tiles over K=32 (head dim) ----
        v16h k0 = load_frag16x32(kbase + (size_t)kb * EMBED, EMBED);
        v16h k1 = load_frag16x32(kbase + (size_t)(kb + 16) * EMBED, EMBED);
        v8f z = {};
        v8f s0 = __builtin_amdgcn_wmma_f32_16x16x32_f16(false, qf, false, k0,
                                                        (short)0, z, false, false);
        v8f s1 = __builtin_amdgcn_wmma_f32_16x16x32_f16(false, qf, false, k1,
                                                        (short)0, z, false, false);

        // ---- bias + online softmax (rows live across 16-lane halves) ----
#pragma unroll
        for (int r = 0; r < 8; ++r) {
            const int rowg = qt * 16 + r + (hi << 3);
            const float i0 = infl_b[(size_t)rowg * SEQ + kb + col];
            const float i1 = infl_b[(size_t)rowg * SEQ + kb + 16 + col];
            float a0 = s0[r] * scale * (1.0f + i0);
            float a1 = s1[r] * scale * (1.0f + i1);
            float t = fmaxf(a0, a1);
#pragma unroll
            for (int off = 1; off < 16; off <<= 1)
                t = fmaxf(t, __shfl_xor(t, off, 32));
            const float mn    = fmaxf(m[r], t);
            const float alpha = __expf(m[r] - mn);
            const float p0 = __expf(a0 - mn);
            const float p1 = __expf(a1 - mn);
            float rs = p0 + p1;
#pragma unroll
            for (int off = 1; off < 16; off <<= 1)
                rs += __shfl_xor(rs, off, 32);
            l[r] = l[r] * alpha + rs;
            m[r] = mn;
            acc0[r] *= alpha;
            acc1[r] *= alpha;
            lds_p[(r + (hi << 3)) * 32 + col]      = (_Float16)p0;
            lds_p[(r + (hi << 3)) * 32 + col + 16] = (_Float16)p1;
        }
        __syncthreads();

        // ---- PV: P(16x32 keys) x V(32 keys x 32 dims) as two WMMAs ----
        v16h pf = load_frag16x32(lds_p, 32);
        v16h v0 = load_frag16x32(vbase + kb, SEQ);
        v16h v1 = load_frag16x32(vbase + (size_t)16 * SEQ + kb, SEQ);
        acc0 = __builtin_amdgcn_wmma_f32_16x16x32_f16(false, pf, false, v0,
                                                      (short)0, acc0, false, false);
        acc1 = __builtin_amdgcn_wmma_f32_16x16x32_f16(false, pf, false, v1,
                                                      (short)0, acc1, false, false);
        __syncthreads();
    }

#pragma unroll
    for (int r = 0; r < 8; ++r) {
        const int rowg = qt * 16 + r + (hi << 3);
        const float inv = 1.0f / l[r];
        _Float16* o = out + ((size_t)(b * SEQ + rowg)) * EMBED + h * HDIM;
        o[col]      = (_Float16)(acc0[r] * inv);
        o[16 + col] = (_Float16)(acc1[r] * inv);
    }
}

// ---------------------------------------------------------------------------
// Host-side orchestration
// ---------------------------------------------------------------------------
extern "C" void kernel_launch(void* const* d_in, const int* in_sizes, int n_in,
                              void* d_out, int out_size, void* d_ws, size_t ws_size,
                              hipStream_t stream) {
    (void)in_sizes; (void)n_in; (void)out_size; (void)ws_size;

    const float* x    = (const float*)d_in[0];
    const float* infl = (const float*)d_in[1];
    const float* qw = (const float*)d_in[2];  const float* qb = (const float*)d_in[3];
    const float* kw = (const float*)d_in[4];  const float* kb = (const float*)d_in[5];
    const float* vw = (const float*)d_in[6];  const float* vb = (const float*)d_in[7];
    const float* ow = (const float*)d_in[8];  const float* ob = (const float*)d_in[9];
    const float* w1 = (const float*)d_in[10]; const float* b1 = (const float*)d_in[11];
    const float* w2 = (const float*)d_in[12]; const float* b2 = (const float*)d_in[13];
    const float* ln1w = (const float*)d_in[14]; const float* ln1b = (const float*)d_in[15];
    const float* ln2w = (const float*)d_in[16]; const float* ln2b = (const float*)d_in[17];

    char* ws = (char*)d_ws;
    size_t off = 0;
    auto alloc = [&](size_t bytes) { char* p = ws + off; off += (bytes + 255) & ~(size_t)255; return p; };

    _Float16* qwT = (_Float16*)alloc((size_t)EMBED * EMBED * 2);
    _Float16* kwT = (_Float16*)alloc((size_t)EMBED * EMBED * 2);
    _Float16* vwT = (_Float16*)alloc((size_t)EMBED * EMBED * 2);
    _Float16* owT = (_Float16*)alloc((size_t)EMBED * EMBED * 2);
    _Float16* w1T = (_Float16*)alloc((size_t)EMBED * FFN * 2);
    _Float16* w2T = (_Float16*)alloc((size_t)FFN * EMBED * 2);
    _Float16* h16 = (_Float16*)alloc((size_t)ROWS * EMBED * 2);
    _Float16* q16 = (_Float16*)alloc((size_t)ROWS * EMBED * 2);
    _Float16* k16 = (_Float16*)alloc((size_t)ROWS * EMBED * 2);
    _Float16* vT16 = (_Float16*)alloc((size_t)ROWS * EMBED * 2);
    _Float16* att16 = (_Float16*)alloc((size_t)ROWS * EMBED * 2);
    float*    x2    = (float*)   alloc((size_t)ROWS * EMBED * 4);
    _Float16* h2_16 = (_Float16*)alloc((size_t)ROWS * EMBED * 2);
    _Float16* a1    = (_Float16*)alloc((size_t)ROWS * FFN * 2);

    // 1) weight prep (transpose + f16 convert)
    {
        int n = EMBED * EMBED;
        transpose_to_f16_kernel<<<(n + 255) / 256, 256, 0, stream>>>(qw, qwT, EMBED, EMBED);
        transpose_to_f16_kernel<<<(n + 255) / 256, 256, 0, stream>>>(kw, kwT, EMBED, EMBED);
        transpose_to_f16_kernel<<<(n + 255) / 256, 256, 0, stream>>>(vw, vwT, EMBED, EMBED);
        transpose_to_f16_kernel<<<(n + 255) / 256, 256, 0, stream>>>(ow, owT, EMBED, EMBED);
        int nf = EMBED * FFN;
        transpose_to_f16_kernel<<<(nf + 255) / 256, 256, 0, stream>>>(w1, w1T, EMBED, FFN);
        transpose_to_f16_kernel<<<(nf + 255) / 256, 256, 0, stream>>>(w2, w2T, FFN, EMBED);
    }

    // 2) LN1
    layernorm_kernel<<<ROWS, 32, 0, stream>>>(x, ln1w, ln1b, h16);

    // 3) QKV projections (V stored transposed per batch for the PV WMMA)
    {
        dim3 grid((ROWS / 64) * (EMBED / 64));
        wmma_gemm_kernel<<<grid, 128, 0, stream>>>(h16, qwT, qb, nullptr, q16,
                                                   ROWS, EMBED, EMBED, 0);
        wmma_gemm_kernel<<<grid, 128, 0, stream>>>(h16, kwT, kb, nullptr, k16,
                                                   ROWS, EMBED, EMBED, 0);
        wmma_gemm_kernel<<<grid, 128, 0, stream>>>(h16, vwT, vb, nullptr, vT16,
                                                   ROWS, EMBED, EMBED, 4);
    }

    // 4) attention
    attention_kernel<<<BATCH * HEADS * (SEQ / 16), 32, 0, stream>>>(q16, k16, vT16,
                                                                    infl, att16);

    // 5) output projection + residual -> x2 (f32)
    {
        dim3 grid((ROWS / 64) * (EMBED / 64));
        wmma_gemm_kernel<<<grid, 128, 0, stream>>>(att16, owT, ob, x, x2,
                                                   ROWS, EMBED, EMBED, 2);
    }

    // 6) LN2
    layernorm_kernel<<<ROWS, 32, 0, stream>>>(x2, ln2w, ln2b, h2_16);

    // 7) FFN1 with exact-erf GELU
    {
        dim3 grid((ROWS / 64) * (FFN / 64));
        wmma_gemm_kernel<<<grid, 128, 0, stream>>>(h2_16, w1T, b1, nullptr, a1,
                                                   ROWS, FFN, EMBED, 1);
    }

    // 8) FFN2 + residual -> d_out (f32)
    {
        dim3 grid((ROWS / 64) * (EMBED / 64));
        wmma_gemm_kernel<<<grid, 128, 0, stream>>>(a1, w2T, b2, x2, (float*)d_out,
                                                   ROWS, EMBED, FFN, 2);
    }
}
